// MotionMamba_24816321036294
// MI455X (gfx1250) — matrix operand order repaired and checked by reading, not compiled, not run on hardware
//
#include <hip/hip_runtime.h>

#define EMBED   256
#define D_STATE 16
#define D_INNER 512
#define DT_RANK 16
#define NJ      17
#define NF      243
#define NB      8
#define MTOK    (NB*NF*NJ)   /* 33048 tokens, same for both layouts */

typedef __attribute__((ext_vector_type(16))) __bf16 v16bf;
typedef __attribute__((ext_vector_type(2)))  __bf16 bf16x2;
typedef __attribute__((ext_vector_type(8)))  float  v8f;

union FragBF { uint4 q[2]; v16bf v; };

// packed f32 pair -> bf16x2 dword: lowers to a single v_cvt_pk_bf16_f32
__device__ __forceinline__ unsigned pk2bf(float x, float y) {
  union { bf16x2 h; unsigned u; } c;
  c.h = (bf16x2){ (__bf16)x, (__bf16)y };
  return c.u;
}
__device__ __forceinline__ unsigned short f2bf(float f) {
  union { __bf16 h; unsigned short u; } c;
  c.h = (__bf16)f;
  return c.u;
}
__device__ __forceinline__ float siluf(float x){ return x / (1.0f + __expf(-x)); }
__device__ __forceinline__ float geluf(float x){ return 0.5f * x * (1.0f + erff(x * 0.70710678118654752f)); }
__device__ __forceinline__ float softplusf(float x){ return (x > 20.0f) ? x : log1pf(__expf(x)); }

// ---------------------------------------------------------------------------
// WMMA GEMM: C[M,N] = A[M,K](lda) * W[N,K]^T  (+bias, epilogue)
// ep: 0=none 1=bias 2=bias+gelu 3=bias+softplus
// Block = 128 threads (4 waves). Block tile = 64(M) x 64(N).
// Each wave owns one 16-wide N strip and reuses its B fragment across the
// four 16-row M subtiles -> 4 back-to-back v_wmma per K step.
// A staged fp32->bf16 into LDS; W pre-converted bf16 (N,K) row-major.
// Requires: K multiple of 16, N multiple of 16, lda >= K, 16B-aligned ptrs.
// ---------------------------------------------------------------------------
__global__ __launch_bounds__(128) void gemm_wmma(
    const float* __restrict__ A, const unsigned short* __restrict__ W,
    const float* __restrict__ bias, float* __restrict__ C,
    int M, int N, int K, int lda, int ep)
{
  __shared__ __align__(16) unsigned short lA[64*32];
  const int m0   = blockIdx.x * 64;
  const int wave = threadIdx.x >> 5;
  const int lane = threadIdx.x & 31;
  const int n0   = blockIdx.y * 64 + wave * 16;
  const int hi   = lane >> 4;
  const int ln   = lane & 15;
  const bool active = (n0 < N);            // wave-uniform -> EXEC all ones inside

  v8f acc[4];
  #pragma unroll
  for (int mt = 0; mt < 4; ++mt) acc[mt] = (v8f){0.f,0.f,0.f,0.f,0.f,0.f,0.f,0.f};

  const int scol = (threadIdx.x & 7) << 2;       // staging column (0..28, step 4)
  const int srow = threadIdx.x >> 3;             // staging row base (0..15)

  for (int k0 = 0; k0 < K; k0 += 32) {
    __syncthreads();
    // stage 64x32 A panel: 128 threads x 4 float4 loads, f32 -> bf16
    // (global_load_b128 -> 2x v_cvt_pk_bf16_f32 -> ds_store_b64).
    // K is a multiple of 16 and scol a multiple of 4, so (gk<K) covers the
    // whole float4 -- no per-element tail path.
    #pragma unroll
    for (int r = 0; r < 4; ++r) {
      int row = srow + 16*r;
      int gm = m0 + row;
      int gk = k0 + scol;
      uint2 sv = make_uint2(0u, 0u);
      if (gm < M && gk < K) {
        float4 f = *(const float4*)(A + (size_t)gm * lda + gk);
        sv = make_uint2(pk2bf(f.x, f.y), pk2bf(f.z, f.w));
      }
      *(uint2*)&lA[row*32 + scol] = sv;
    }
    __syncthreads();
    if (active) {
      // B 32x16 bf16 fragment: lane holds column N=n0+ln, K = k0+hi*16 .. +15
      FragBF b;
      int kk = k0 + (hi << 4);
      if (kk < K) {
        const unsigned short* wp = W + (size_t)(n0 + ln) * K + kk;
        b.q[0] = ((const uint4*)wp)[0];
        b.q[1] = ((const uint4*)wp)[1];
      } else {
        b.q[0] = make_uint4(0,0,0,0); b.q[1] = make_uint4(0,0,0,0);
      }
      // A 16x32 bf16 fragments: lane ln holds row M=ln;
      // halfs 0-15 = K[hi*16..+15].
      // Issue all 8 ds_load_b128 first, then 4 back-to-back WMMAs.
      FragBF a[4];
      #pragma unroll
      for (int mt = 0; mt < 4; ++mt) {
        int base = ((mt*16 + ln) << 5) + (hi << 4);
        a[mt].q[0] = *(const uint4*)&lA[base];
        a[mt].q[1] = *(const uint4*)&lA[base + 8];
      }
      #pragma unroll
      for (int mt = 0; mt < 4; ++mt) {
        acc[mt] = __builtin_amdgcn_wmma_f32_16x16x32_bf16(
            false, a[mt].v, false, b.v, (short)0, acc[mt], false, false);
      }
    }
  }

  if (active) {
    int n = n0 + ln;
    float bv = bias ? bias[n] : 0.0f;
    #pragma unroll
    for (int mt = 0; mt < 4; ++mt) {
      #pragma unroll
      for (int r = 0; r < 8; ++r) {        // D layout: VGPR r -> M = r + hi*8
        int m = m0 + mt*16 + r + hi * 8;
        if (m < M) {
          float v = acc[mt][r] + bv;
          if (ep == 2) v = geluf(v);
          else if (ep == 3) v = softplusf(v);
          C[(size_t)m * N + n] = v;
        }
      }
    }
  }
}

// ---------------------------------------------------------------------------
__global__ void cvt_bf16(const float* __restrict__ s, unsigned short* __restrict__ d, int n){
  int i = blockIdx.x * 256 + threadIdx.x;
  if (i < n) d[i] = f2bf(s[i]);
}

__global__ void embed_kernel(const float* __restrict__ x, const float* __restrict__ ew,
                             const float* __restrict__ eb, const float* __restrict__ spos,
                             float* __restrict__ X){
  int tok = blockIdx.x, e = threadIdx.x;
  int n = tok % NJ;
  float x0 = x[tok*3+0], x1 = x[tok*3+1], x2 = x[tok*3+2];
  X[(size_t)tok*EMBED + e] =
      x0*ew[e*3+0] + x1*ew[e*3+1] + x2*ew[e*3+2] + eb[e] + spos[n*EMBED + e];
}

__global__ __launch_bounds__(256) void ln_kernel(
    const float* __restrict__ X, const float* __restrict__ g,
    const float* __restrict__ b, float* __restrict__ O, int M){
  int wave = threadIdx.x >> 5, lane = threadIdx.x & 31;
  int tok = blockIdx.x * 8 + wave;
  if (tok >= M) return;
  const float* x = X + (size_t)tok * EMBED;
  float v[8], s = 0.f, s2 = 0.f;
  #pragma unroll
  for (int i = 0; i < 8; ++i) { float t = x[lane + i*32]; v[i] = t; s += t; s2 += t*t; }
  for (int o = 16; o > 0; o >>= 1) { s += __shfl_xor(s, o, 32); s2 += __shfl_xor(s2, o, 32); }
  float mean = s * (1.f/256.f);
  float var  = s2 * (1.f/256.f) - mean*mean;
  float inv  = rsqrtf(var + 1e-6f);
  float* out = O + (size_t)tok * EMBED;
  #pragma unroll
  for (int i = 0; i < 8; ++i) { int e = lane + i*32; out[e] = (v[i]-mean)*inv*g[e] + b[e]; }
}

// causal depthwise conv (k=4) + SiLU ; reads xc half of XZ (stride 1024)
__global__ void conv_silu_kernel(const float* __restrict__ XZ, const float* __restrict__ cw,
                                 const float* __restrict__ cb, float* __restrict__ OUT,
                                 int S, int L){
  int id = blockIdx.x * blockDim.x + threadIdx.x;
  if (id >= S * D_INNER) return;
  int ch = id & (D_INNER - 1), seq = id >> 9;
  float w0 = cw[ch*4], w1 = cw[ch*4+1], w2 = cw[ch*4+2], w3 = cw[ch*4+3], bb = cb[ch];
  const float* px = XZ  + (size_t)seq * L * (2*D_INNER) + ch;
  float*       po = OUT + (size_t)seq * L * D_INNER     + ch;
  float xm3 = 0.f, xm2 = 0.f, xm1 = 0.f;
  for (int l = 0; l < L; ++l) {
    float x0 = px[(size_t)l * (2*D_INNER)];
    po[(size_t)l * D_INNER] = siluf(bb + w0*xm3 + w1*xm2 + w2*xm1 + w3*x0);
    xm3 = xm2; xm2 = xm1; xm1 = x0;
  }
}

// selective scan + skip (u*D) + SiLU gate; one thread per (seq, channel)
__global__ void scan_kernel(const float* __restrict__ DT, const float* __restrict__ U,
                            const float* __restrict__ DBL, const float* __restrict__ XZ,
                            const float* __restrict__ Alog, const float* __restrict__ Dp,
                            float* __restrict__ Y, int S, int L){
  int id = blockIdx.x * blockDim.x + threadIdx.x;
  if (id >= S * D_INNER) return;
  int ch = id & (D_INNER - 1), seq = id >> 9;
  float a[D_STATE], h[D_STATE];
  #pragma unroll
  for (int s = 0; s < D_STATE; ++s) { a[s] = -__expf(Alog[ch*D_STATE + s]); h[s] = 0.f; }
  float Dch = Dp[ch];
  for (int l = 0; l < L; ++l) {
    size_t tok = (size_t)seq * L + l;
    float dt = DT[tok * D_INNER + ch];
    float u  = U [tok * D_INNER + ch];
    const float* row = DBL + tok * 48;
    float du = dt * u, y = 0.f;
    #pragma unroll
    for (int s = 0; s < D_STATE; ++s) {
      h[s] = __expf(dt * a[s]) * h[s] + du * row[DT_RANK + s];
      y += h[s] * row[DT_RANK + D_STATE + s];
    }
    float z = XZ[tok * (2*D_INNER) + D_INNER + ch];
    Y[tok * D_INNER + ch] = (y + u * Dch) * siluf(z);
  }
}

__global__ void add_kernel(float* __restrict__ P, const float* __restrict__ Q, int n){
  int i = blockIdx.x * 256 + threadIdx.x;
  if (i < n) P[i] += Q[i];
}

__global__ void s2t_kernel(const float* __restrict__ Sb, const float* __restrict__ tpos,
                           float* __restrict__ Tb, int addpos){
  int tok = blockIdx.x, e = threadIdx.x;           // tok in temporal order
  int f = tok % NF; int bn = tok / NF; int n = bn % NJ; int b = bn / NJ;
  int src = (b*NF + f)*NJ + n;
  float v = Sb[(size_t)src*EMBED + e];
  if (addpos) v += tpos[f*EMBED + e];
  Tb[(size_t)tok*EMBED + e] = v;
}

__global__ void t2s_kernel(const float* __restrict__ Tb, float* __restrict__ Sb){
  int tok = blockIdx.x, e = threadIdx.x;           // tok in spatial order
  int n = tok % NJ; int bf = tok / NJ; int f = bf % NF; int b = bf / NF;
  int src = (b*NJ + n)*NF + f;
  Sb[(size_t)tok*EMBED + e] = Tb[(size_t)src*EMBED + e];
}

__global__ __launch_bounds__(256) void head_kernel(
    const float* __restrict__ X, const float* __restrict__ hw,
    const float* __restrict__ hb, float* __restrict__ out, int M){
  int wave = threadIdx.x >> 5, lane = threadIdx.x & 31;
  int tok = blockIdx.x * 8 + wave;
  if (tok >= M) return;
  const float* x = X + (size_t)tok * EMBED;
  for (int j = 0; j < 3; ++j) {
    float s = 0.f;
    #pragma unroll
    for (int i = 0; i < 8; ++i) { int e = lane + i*32; s += x[e] * hw[j*EMBED + e]; }
    for (int o = 16; o > 0; o >>= 1) s += __shfl_xor(s, o, 32);
    if (lane == 0) out[tok*3 + j] = s + hb[j];
  }
}

// ---------------------------------------------------------------------------
extern "C" void kernel_launch(void* const* d_in, const int* in_sizes, int n_in,
                              void* d_out, int out_size, void* d_ws, size_t ws_size,
                              hipStream_t stream) {
  (void)in_sizes; (void)n_in; (void)out_size; (void)ws_size;
  auto F = [&](int i) -> const float* { return (const float*)d_in[i]; };
  const int M = MTOK;

  // workspace layout (fp32 activations + bf16 weight arena)
  float* ws   = (float*)d_ws;
  float* X    = ws;
  float* T1   = X    + (size_t)M*EMBED;
  float* T2   = T1   + (size_t)M*EMBED;
  float* XT   = T2   + (size_t)M*EMBED;
  float* XZ   = XT   + (size_t)M*EMBED;
  float* CONV = XZ   + (size_t)M*2*D_INNER;
  float* DBL  = CONV + (size_t)M*D_INNER;
  float* DTb  = DBL  + (size_t)M*48;
  float* Y5   = DTb  + (size_t)M*D_INNER;
  unsigned short* WBF = (unsigned short*)(Y5 + (size_t)M*D_INNER);

  const size_t WBLK = 688128;  // bf16 elements per block's GEMM weights
  const size_t OFF_IN = 0, OFF_XP = 262144, OFF_DT = 286720,
               OFF_OUT = 294912, OFF_FC1 = 425984, OFF_FC2 = 557056;

  auto cvt = [&](const float* s, unsigned short* d, int n){
    cvt_bf16<<<(n + 255)/256, 256, 0, stream>>>(s, d, n);
  };
  // input index map: 0=x 1=embed_w 2=embed_b 3=spos 4=tpos
  // st[i] @ 5+i*17, tt[i] @ 73+i*17 (17 tensors each),
  // 141..146 = sng snb tng tnb hng hnb, 147 head_w, 148 head_b
  for (int blk = 0; blk < 8; ++blk) {
    int layer = blk & 3;
    int base = 5 + ((blk >= 4) ? 68 : 0) + layer*17;
    unsigned short* wb = WBF + (size_t)blk * WBLK;
    cvt(F(base+4),  wb+OFF_IN,  2*D_INNER*EMBED);   // in_w   (1024,256)
    cvt(F(base+7),  wb+OFF_XP,  48*D_INNER);        // xproj  (48,512)
    cvt(F(base+8),  wb+OFF_DT,  D_INNER*DT_RANK);   // dt_w   (512,16)
    cvt(F(base+12), wb+OFF_OUT, EMBED*D_INNER);     // out_w  (256,512)
    cvt(F(base+13), wb+OFF_FC1, 2*EMBED*EMBED);     // fc1_w  (512,256)
    cvt(F(base+15), wb+OFF_FC2, EMBED*2*EMBED);     // fc2_w  (256,512)
  }

  auto gemm = [&](const float* A, const unsigned short* W, const float* bias,
                  float* C, int N, int K, int lda, int ep){
    dim3 g((M + 63)/64, (N + 63)/64);
    gemm_wmma<<<g, 128, 0, stream>>>(A, W, bias, C, M, N, K, lda, ep);
  };
  auto ln = [&](const float* Xp, const float* g, const float* b, float* O){
    ln_kernel<<<(M + 7)/8, 256, 0, stream>>>(Xp, g, b, O, M);
  };
  auto addv = [&](float* P, const float* Q){
    int n = M * EMBED;
    add_kernel<<<(n + 255)/256, 256, 0, stream>>>(P, Q, n);
  };

  auto run_block = [&](float* P, int S, int L, int base, int blk){
    unsigned short* wb = WBF + (size_t)blk * WBLK;
    int sc = S * D_INNER;
    ln(P, F(base+0), F(base+1), T1);
    gemm(T1, wb+OFF_IN, nullptr, XZ, 2*D_INNER, EMBED, EMBED, 0);
    conv_silu_kernel<<<(sc + 255)/256, 256, 0, stream>>>(XZ, F(base+5), F(base+6), CONV, S, L);
    gemm(CONV, wb+OFF_XP, nullptr, DBL, 48, D_INNER, D_INNER, 0);
    gemm(DBL, wb+OFF_DT, F(base+9), DTb, D_INNER, DT_RANK, 48, 3);   // softplus
    scan_kernel<<<(sc + 255)/256, 256, 0, stream>>>(DTb, CONV, DBL, XZ,
                                                    F(base+10), F(base+11), Y5, S, L);
    gemm(Y5, wb+OFF_OUT, nullptr, T2, EMBED, D_INNER, D_INNER, 0);
    addv(P, T2);
    ln(P, F(base+2), F(base+3), T1);
    gemm(T1, wb+OFF_FC1, F(base+14), Y5, 2*EMBED, EMBED, EMBED, 2);  // gelu
    gemm(Y5, wb+OFF_FC2, F(base+16), T2, EMBED, 2*EMBED, 2*EMBED, 1);
    addv(P, T2);
  };

  embed_kernel<<<M, 256, 0, stream>>>(F(0), F(1), F(2), F(3), X);
  for (int layer = 0; layer < 4; ++layer) {
    int sbase = 5 + layer*17;
    int tbase = 73 + layer*17;
    run_block(X, NB*NF, NJ, sbase, layer);          // spatial, L=17
    ln(X, F(141), F(142), T1);
    s2t_kernel<<<M, 256, 0, stream>>>(T1, F(4), XT, layer == 0 ? 1 : 0);
    run_block(XT, NB*NJ, NF, tbase, 4 + layer);     // temporal, L=243
    ln(XT, F(143), F(144), T1);
    t2s_kernel<<<M, 256, 0, stream>>>(T1, X);
  }
  ln(X, F(145), F(146), T1);
  head_kernel<<<(M + 7)/8, 256, 0, stream>>>(T1, F(147), F(148), (float*)d_out, M);
}
